// MultiHeadAttention_63342177681567
// MI455X (gfx1250) — matrix operand order repaired and checked by reading, not compiled, not run on hardware
//
#include <hip/hip_runtime.h>
#include <hip/hip_bf16.h>

typedef __attribute__((ext_vector_type(16))) _Float16 v16h;
typedef __attribute__((ext_vector_type(8)))  float    v8f;

#define BB 4
#define LL 2048
#define DD 512
#define HH 8

// ---------------------------------------------------------------------------
// WMMA helpers (CDNA5 wave32, v_wmma_f32_16x16x32_f16)
// ---------------------------------------------------------------------------
__device__ __forceinline__ v8f wmma_f16(v16h a, v16h b, v8f c) {
  return __builtin_amdgcn_wmma_f32_16x16x32_f16(
      /*neg_a=*/false, a, /*neg_b=*/false, b,
      /*c_mod=*/(short)0, c, /*reuse_a=*/false, /*reuse_b=*/false);
}

// A fragment: 16x32 (MxK) f16 tile stored row-major (stride halves).
// Lane m<16 holds row m, K = {0..7, 16..23}; lane m+16 holds K = {8..15, 24..31}.
__device__ __forceinline__ v16h load_fragA(const _Float16* base, int stride) {
  int lane = threadIdx.x & 31;
  int m = lane & 15, hi = lane >> 4;
  v16h a;
#pragma unroll
  for (int v = 0; v < 8; ++v) {
    int k = 2 * v + (v >= 4 ? 8 : 0) + hi * 8;
    a[2 * v]     = base[m * stride + k];
    a[2 * v + 1] = base[m * stride + k + 1];
  }
  return a;
}

// B fragment: 32x16 (KxN). Source tile stored transposed as (N x K) row-major,
// i.e. base[n*stride + k] == B[k][n]. Lane n<16: K={0..7,16..23}; lane n+16: K={8..15,24..31}.
__device__ __forceinline__ v16h load_fragB(const _Float16* base, int stride) {
  int lane = threadIdx.x & 31;
  int n = lane & 15, hi = lane >> 4;
  v16h b;
#pragma unroll
  for (int v = 0; v < 8; ++v) {
    int k = 2 * v + (v >= 4 ? 8 : 0) + hi * 8;
    b[2 * v]     = base[n * stride + k];
    b[2 * v + 1] = base[n * stride + k + 1];
  }
  return b;
}

// ---------------------------------------------------------------------------
// K1: projection  C = X(8192x512) @ W^T(512x512) + bias, scaled, stored f16 in
// head-major layout out[b][h][l][64].  Block = 256 thr (8 waves), tile 64x64.
// ---------------------------------------------------------------------------
__global__ __launch_bounds__(256) void proj_kernel(
    const float* __restrict__ X, const float* __restrict__ W,
    const float* __restrict__ bias, _Float16* __restrict__ out, float scale) {
  __shared__ _Float16 As[64 * 32];
  __shared__ _Float16 Bs[64 * 32];
  int tid = threadIdx.x;
  int w = tid >> 5, lane = tid & 31;
  int hi = lane >> 4, cn = lane & 15;
  int m0 = blockIdx.x * 64;
  int n0 = blockIdx.y * 64;
  int mt = w & 3;   // wave's 16-row slab
  int nh = w >> 2;  // wave's 32-col half
  v8f acc[2] = {{}, {}};

  for (int k0 = 0; k0 < 512; k0 += 32) {
#pragma unroll
    for (int it = 0; it < 2; ++it) {
      int slot = tid + it * 256;           // 0..511
      int r = slot >> 3;                   // row 0..63
      int c4 = (slot & 7) * 4;             // col group
      float4 fa = *(const float4*)(X + (size_t)(m0 + r) * 512 + k0 + c4);
      As[r * 32 + c4 + 0] = (_Float16)fa.x;
      As[r * 32 + c4 + 1] = (_Float16)fa.y;
      As[r * 32 + c4 + 2] = (_Float16)fa.z;
      As[r * 32 + c4 + 3] = (_Float16)fa.w;
      float4 fb = *(const float4*)(W + (size_t)(n0 + r) * 512 + k0 + c4);
      Bs[r * 32 + c4 + 0] = (_Float16)fb.x;
      Bs[r * 32 + c4 + 1] = (_Float16)fb.y;
      Bs[r * 32 + c4 + 2] = (_Float16)fb.z;
      Bs[r * 32 + c4 + 3] = (_Float16)fb.w;
    }
    __syncthreads();
    v16h a = load_fragA(&As[(16 * mt) * 32], 32);
#pragma unroll
    for (int t = 0; t < 2; ++t) {
      v16h b = load_fragB(&Bs[(32 * nh + 16 * t) * 32], 32);
      acc[t] = wmma_f16(a, b, acc[t]);
    }
    __syncthreads();
  }
#pragma unroll
  for (int t = 0; t < 2; ++t) {
    int ncol = n0 + 32 * nh + 16 * t + cn;
    float bv = bias[ncol];
    int h = ncol >> 6, dk = ncol & 63;
#pragma unroll
    for (int g = 0; g < 8; ++g) {
      int mrow = m0 + 16 * mt + g + hi * 8;
      int b = mrow >> 11, l = mrow & 2047;
      float val = (acc[t][g] + bv) * scale;
      out[(((size_t)(b * HH + h)) * LL + l) * 64 + dk] = (_Float16)val;
    }
  }
}

// ---------------------------------------------------------------------------
// K2: attention per (b,h).  Block = 128 thr (4 waves), 64 query rows/block,
// 16 rows/wave.  Two passes over 64 key-chunks of 32:
//   pass1: rowsum of exp(S) (mask -> 0)
//   pass2: recompute S, write normalized attn (fp32), P f16 -> LDS -> A-frag,
//          O += P@V, then scatter O into the reinterpretation-permuted fcin.
// ---------------------------------------------------------------------------
__global__ __launch_bounds__(128) void attn_kernel(
    const _Float16* __restrict__ qh, const _Float16* __restrict__ kh,
    const _Float16* __restrict__ vh, const int* __restrict__ mask,
    float* __restrict__ attn, _Float16* __restrict__ fcin) {
  __shared__ _Float16 Ks[32 * 64];      // (key, dk)  stride 64
  __shared__ _Float16 Vt[64 * 32];      // (dv, key)  stride 32  (transposed)
  __shared__ _Float16 Ps[4][16 * 32];   // per-wave P staging

  int tid = threadIdx.x;
  int w = tid >> 5, lane = tid & 31;
  int hi = lane >> 4, lnn = lane & 15;
  int bh = blockIdx.y;
  int b = bh >> 3;
  int q0 = blockIdx.x * 64 + 16 * w;
  const _Float16* qptr = qh + ((size_t)bh * LL) * 64;
  const _Float16* kptr = kh + ((size_t)bh * LL) * 64;
  const _Float16* vptr = vh + ((size_t)bh * LL) * 64;
  const int* mbase = mask + ((size_t)b * LL) * LL;

  // resident Q A-fragments for dk 0..31 and 32..63 (Q already scaled by 1/8)
  v16h qa0, qa1;
#pragma unroll
  for (int v = 0; v < 8; ++v) {
    int k = 2 * v + (v >= 4 ? 8 : 0) + hi * 8;
    const _Float16* p = qptr + (size_t)(q0 + lnn) * 64;
    qa0[2 * v] = p[k];      qa0[2 * v + 1] = p[k + 1];
    qa1[2 * v] = p[32 + k]; qa1[2 * v + 1] = p[32 + k + 1];
  }

  float psum[8];
#pragma unroll
  for (int g = 0; g < 8; ++g) psum[g] = 0.f;

  // ---- pass 1: row sums ----
  for (int c = 0; c < 64; ++c) {
    int kb = c * 32;
#pragma unroll
    for (int it = 0; it < 2; ++it) {
      int slot = tid + it * 128;
      int r = slot >> 3, c8 = (slot & 7) * 8;
      *(uint4*)&Ks[r * 64 + c8] = *(const uint4*)&kptr[(size_t)(kb + r) * 64 + c8];
    }
    __syncthreads();
#pragma unroll
    for (int t = 0; t < 2; ++t) {
      v8f s = {};
      s = wmma_f16(qa0, load_fragB(&Ks[(16 * t) * 64], 64), s);
      s = wmma_f16(qa1, load_fragB(&Ks[(16 * t) * 64 + 32], 64), s);
      int key = kb + 16 * t + lnn;
#pragma unroll
      for (int g = 0; g < 8; ++g) {
        int row = q0 + g + hi * 8;
        int mv = mbase[(size_t)row * LL + key];
        psum[g] += (mv == 1) ? 0.f : __expf(s[g]);
      }
    }
    __syncthreads();
  }
#pragma unroll
  for (int g = 0; g < 8; ++g) {
#pragma unroll
    for (int off = 1; off < 16; off <<= 1)
      psum[g] += __shfl_xor(psum[g], off, 32);
    psum[g] = 1.0f / psum[g];
  }

  // ---- pass 2 ----
  v8f oacc[4] = {{}, {}, {}, {}};
  for (int c = 0; c < 64; ++c) {
    int kb = c * 32;
#pragma unroll
    for (int it = 0; it < 2; ++it) {
      int slot = tid + it * 128;
      int r = slot >> 3, c8 = (slot & 7) * 8;
      *(uint4*)&Ks[r * 64 + c8] = *(const uint4*)&kptr[(size_t)(kb + r) * 64 + c8];
      uint4 vv = *(const uint4*)&vptr[(size_t)(kb + r) * 64 + c8];
      const _Float16* v8 = (const _Float16*)&vv;
#pragma unroll
      for (int z = 0; z < 8; ++z) Vt[(c8 + z) * 32 + r] = v8[z];
    }
    __syncthreads();
#pragma unroll
    for (int t = 0; t < 2; ++t) {
      v8f s = {};
      s = wmma_f16(qa0, load_fragB(&Ks[(16 * t) * 64], 64), s);
      s = wmma_f16(qa1, load_fragB(&Ks[(16 * t) * 64 + 32], 64), s);
      int key = kb + 16 * t + lnn;
#pragma unroll
      for (int g = 0; g < 8; ++g) {
        int row = q0 + g + hi * 8;
        int mv = mbase[(size_t)row * LL + key];
        float e = (mv == 1) ? 0.f : __expf(s[g]) * psum[g];
        attn[((size_t)bh * LL + row) * LL + key] = e;
        Ps[w][(g + hi * 8) * 32 + 16 * t + lnn] = (_Float16)e;
      }
    }
    // wave-local LDS RAW (DS ops are in-order within a wave)
    v16h pa = load_fragA(&Ps[w][0], 32);
#pragma unroll
    for (int j = 0; j < 4; ++j) {
      v16h bv = load_fragB(&Vt[(16 * j) * 32], 32);
      oacc[j] = wmma_f16(pa, bv, oacc[j]);
    }
    __syncthreads();
  }

  // scatter O through the faithful (B,H,L,DV)->view(H,B,L,DV)->perm mapping
  size_t pbase = ((size_t)(bh >> 2)) * 524288 + ((size_t)(bh & 3)) * 64;
#pragma unroll
  for (int j = 0; j < 4; ++j) {
#pragma unroll
    for (int g = 0; g < 8; ++g) {
      int l = q0 + g + hi * 8;
      int dv = 16 * j + lnn;
      fcin[pbase + (size_t)l * 256 + dv] = (_Float16)oacc[j][g];
    }
  }
}

// ---------------------------------------------------------------------------
// fp32 -> f16 weight convert
// ---------------------------------------------------------------------------
__global__ void cvt_f16(const float* __restrict__ src, _Float16* __restrict__ dst,
                        int n) {
  int i = blockIdx.x * 256 + threadIdx.x;
  if (i < n) dst[i] = (_Float16)src[i];
}

// ---------------------------------------------------------------------------
// K3: y = LN( fcin @ fc_w^T + fc_b + residual ).  Block = 256 thr (8 waves),
// 16 rows x full 512 cols per block; cross-wave LN reduction via LDS.
// ---------------------------------------------------------------------------
__global__ __launch_bounds__(256) void fc_ln_kernel(
    const _Float16* __restrict__ fcin, const _Float16* __restrict__ fcw16,
    const float* __restrict__ fc_b, const float* __restrict__ resid,
    const float* __restrict__ ln_g, const float* __restrict__ ln_b,
    float* __restrict__ y) {
  __shared__ float s_sum[16][8];
  __shared__ float s_sq[16][8];
  __shared__ float s_mu[16];
  __shared__ float s_rs[16];
  int tid = threadIdx.x;
  int w = tid >> 5, lane = tid & 31;
  int hi = lane >> 4, lnn = lane & 15;
  int m0 = blockIdx.x * 16;

  v8f acc[4] = {{}, {}, {}, {}};
  for (int k0 = 0; k0 < 512; k0 += 32) {
    v16h a;
#pragma unroll
    for (int v = 0; v < 8; ++v) {
      int k = 2 * v + (v >= 4 ? 8 : 0) + hi * 8;
      const _Float16* p = fcin + (size_t)(m0 + lnn) * 512 + k0 + k;
      a[2 * v] = p[0];
      a[2 * v + 1] = p[1];
    }
#pragma unroll
    for (int j = 0; j < 4; ++j) {
      int nrow = 64 * w + 16 * j + lnn;
      v16h bf;
#pragma unroll
      for (int v = 0; v < 8; ++v) {
        int k = 2 * v + (v >= 4 ? 8 : 0) + hi * 8;
        const _Float16* p = fcw16 + (size_t)nrow * 512 + k0 + k;
        bf[2 * v] = p[0];
        bf[2 * v + 1] = p[1];
      }
      acc[j] = wmma_f16(a, bf, acc[j]);
    }
  }

  float lsum[8], lsq[8];
#pragma unroll
  for (int g = 0; g < 8; ++g) { lsum[g] = 0.f; lsq[g] = 0.f; }
#pragma unroll
  for (int j = 0; j < 4; ++j) {
    int col = 64 * w + 16 * j + lnn;
    float bb = fc_b[col];
#pragma unroll
    for (int g = 0; g < 8; ++g) {
      int M = m0 + g + hi * 8;
      float x = acc[j][g] + bb + resid[(size_t)M * 512 + col];
      acc[j][g] = x;
      lsum[g] += x;
      lsq[g] += x * x;
    }
  }
#pragma unroll
  for (int g = 0; g < 8; ++g) {
#pragma unroll
    for (int off = 1; off < 16; off <<= 1) {
      lsum[g] += __shfl_xor(lsum[g], off, 32);
      lsq[g]  += __shfl_xor(lsq[g], off, 32);
    }
  }
  if (lnn == 0) {
#pragma unroll
    for (int g = 0; g < 8; ++g) {
      s_sum[g + hi * 8][w] = lsum[g];
      s_sq[g + hi * 8][w]  = lsq[g];
    }
  }
  __syncthreads();
  if (tid < 16) {
    float s = 0.f, q2 = 0.f;
#pragma unroll
    for (int ww = 0; ww < 8; ++ww) { s += s_sum[tid][ww]; q2 += s_sq[tid][ww]; }
    float mu = s * (1.0f / 512.0f);
    float var = q2 * (1.0f / 512.0f) - mu * mu;
    s_mu[tid] = mu;
    s_rs[tid] = rsqrtf(var + 1e-5f);
  }
  __syncthreads();
#pragma unroll
  for (int j = 0; j < 4; ++j) {
    int col = 64 * w + 16 * j + lnn;
    float gg = ln_g[col], bb = ln_b[col];
#pragma unroll
    for (int g = 0; g < 8; ++g) {
      int r = g + hi * 8;
      float yv = (acc[j][g] - s_mu[r]) * s_rs[r] * gg + bb;
      y[(size_t)(m0 + r) * 512 + col] = yv;
    }
  }
}

// ---------------------------------------------------------------------------
extern "C" void kernel_launch(void* const* d_in, const int* in_sizes, int n_in,
                              void* d_out, int out_size, void* d_ws,
                              size_t ws_size, hipStream_t stream) {
  const float* q    = (const float*)d_in[0];
  const float* k    = (const float*)d_in[1];
  const float* v    = (const float*)d_in[2];
  const int*   mask = (const int*)d_in[3];
  const float* wq_w = (const float*)d_in[4];
  const float* wq_b = (const float*)d_in[5];
  const float* wv_w = (const float*)d_in[6];
  const float* wv_b = (const float*)d_in[7];
  const float* fc_w = (const float*)d_in[8];
  const float* fc_b = (const float*)d_in[9];
  const float* ln_g = (const float*)d_in[10];
  const float* ln_b = (const float*)d_in[11];

  float* y    = (float*)d_out;
  float* attn = y + (size_t)BB * LL * DD;   // 4,194,304 floats of y first

  char* ws = (char*)d_ws;
  _Float16* qh    = (_Float16*)(ws);
  _Float16* kh    = (_Float16*)(ws + ((size_t)8 << 20));
  _Float16* vh    = (_Float16*)(ws + ((size_t)16 << 20));
  _Float16* fcin  = (_Float16*)(ws + ((size_t)24 << 20));
  _Float16* fcw16 = (_Float16*)(ws + ((size_t)32 << 20));

  dim3 gP(128, 8);
  proj_kernel<<<gP, 256, 0, stream>>>(q, wq_w, wq_b, qh, 0.125f);  // Q/TEMP folded
  proj_kernel<<<gP, 256, 0, stream>>>(k, wq_w, wq_b, kh, 1.0f);    // bug: K uses wq
  proj_kernel<<<gP, 256, 0, stream>>>(v, wv_w, wv_b, vh, 1.0f);
  cvt_f16<<<(512 * 512 + 255) / 256, 256, 0, stream>>>(fc_w, fcw16, 512 * 512);
  attn_kernel<<<dim3(32, 32), 128, 0, stream>>>(qh, kh, vh, mask, attn, fcin);
  fc_ln_kernel<<<512, 256, 0, stream>>>(fcin, fcw16, fc_b, q, ln_g, ln_b, y);
}
